// TransformerModel_16544214024281
// MI455X (gfx1250) — compile-verified, matrix operand order checked by reading, not run on hardware
//
#include <hip/hip_runtime.h>
#include <hip/hip_bf16.h>

typedef __attribute__((ext_vector_type(16))) _Float16 v16h;
typedef __attribute__((ext_vector_type(8)))  _Float16 v8h;
typedef __attribute__((ext_vector_type(8)))  float    v8f;

#define BM 128
#define BK 32

// ---------------------------------------------------------------------------
// Async global->LDS staging of a ROWS x 32 f16 tile (K-major rows of 64B).
// Each lane issues 16B copies tracked by ASYNCcnt (CDNA5 async-tensor path).
// ---------------------------------------------------------------------------
template <int ROWS>
__device__ __forceinline__ void stage_async(_Float16* dst, const _Float16* src,
                                            int ld, int tid)
{
    const unsigned base = (unsigned)(uintptr_t)(void*)dst;  // low 32b = LDS offset
#pragma unroll
    for (int i = 0; i < (ROWS * 4) / 256; ++i) {
        const int e   = i * 256 + tid;
        const int row = e >> 2;
        const int ch  = e & 3;                  // 16B chunk within 64B row
        const void* g = (const void*)(src + (long)row * ld + ch * 8);
        const unsigned l = base + (unsigned)((row * BK + ch * 8) * 2);
        asm volatile("global_load_async_to_lds_b128 %0, %1, off"
                     :: "v"(l), "v"(g) : "memory");
    }
}

__device__ __forceinline__ void wait_async0()
{
    asm volatile("s_wait_asynccnt 0" ::: "memory");
}

// ---------------------------------------------------------------------------
// WMMA GEMM: C[m,n] = alpha * sum_k A[m,k]*B(n,k) + bias[n]
//  TBN : 128 (8 waves as 2x4) or 64 (8 waves as 4x2, for N=64)
//  AH  : A is f16 (async-staged) else f32 (convert-staged)
//  BT  : B stored [N,K] (f16, async-staged) else [K,N] (f16, transpose-staged)
//  CH  : C written as f16 else f32
// Double-buffered LDS: tile k+1's async DMA overlaps tile k's WMMAs; one
// barrier per K-step. All dims exact multiples of the tile (no guards).
// ---------------------------------------------------------------------------
template <int TBN, bool AH, bool BT, bool RELU, bool CH>
__global__ __launch_bounds__(256) void wmma_gemm(
    const void* __restrict__ Av, int lda, long sAb, long sAh,
    const void* __restrict__ Bv, int ldb, long sBb, long sBh,
    const float* __restrict__ bias,
    void* __restrict__ Cv, int ldc, long sCb, long sCh,
    int M, int N, int K, float alpha, int zH)
{
    constexpr int WNW = TBN / 32;          // waves along N (4 or 2)
    constexpr int WMW = 8 / WNW;           // waves along M (2 or 4)
    constexpr int MF  = (BM / WMW) / 16;   // 4 or 2 m-fragments per wave
    constexpr int NF  = 2;                 // 2 n-fragments per wave
    constexpr bool ASY = AH || BT;

    __shared__ alignas(16) _Float16 As[2][BM * BK];
    __shared__ alignas(16) _Float16 Bs[2][TBN * BK];

    const int zb = blockIdx.z / zH, zh = blockIdx.z % zH;
    const long offA = (long)zb * sAb + (long)zh * sAh;
    const long offB = (long)zb * sBb + (long)zh * sBh;
    const long offC = (long)zb * sCb + (long)zh * sCh;

    const float*    Af  = AH ? nullptr : ((const float*)Av + offA);
    const _Float16* Ahp = AH ? ((const _Float16*)Av + offA) : nullptr;
    const _Float16* Bh  = (const _Float16*)Bv + offB;
    float*    Cf  = CH ? nullptr : ((float*)Cv + offC);
    _Float16* Chp = CH ? ((_Float16*)Cv + offC) : nullptr;

    const int tile_n = blockIdx.x * TBN;
    const int tile_m = blockIdx.y * BM;

    const int tid  = threadIdx.x;
    const int lane = tid & 31;
    const int wave = tid >> 5;
    const int w_m0 = (wave / WNW) * (MF * 16);
    const int w_n0 = (wave % WNW) * 32;
    const int lane16  = lane & 15;
    const int laneHi8 = (lane >> 4) * 8;

    // ---- tile staging helpers ----
    auto stageA = [&](int buf, int k0) {
        if (AH) {
            stage_async<BM>(As[buf], Ahp + (long)tile_m * lda + k0, lda, tid);
        } else {
            const int row = tid >> 1;
            const int kc  = (tid & 1) * 16;
            const float* srcp = Af + (long)(tile_m + row) * lda + k0 + kc;
            __builtin_prefetch(srcp + BK, 0, 1);   // tile k+2 stream hint
            v8h h0, h1;
#pragma unroll
            for (int i = 0; i < 8; ++i) h0[i] = (_Float16)srcp[i];
#pragma unroll
            for (int i = 0; i < 8; ++i) h1[i] = (_Float16)srcp[8 + i];
            *(v8h*)&As[buf][row * BK + kc]     = h0;
            *(v8h*)&As[buf][row * BK + kc + 8] = h1;
        }
    };
    auto stageB = [&](int buf, int k0) {
        if (BT) {
            stage_async<TBN>(Bs[buf], Bh + (long)tile_n * ldb + k0, ldb, tid);
        } else {
            // B is [K, N] f16: transpose into K-major LDS layout (TBN=64)
            const int kk = tid >> 3;            // 0..31
            const int nc = (tid & 7) * 8;       // 0..56
            const _Float16* srcp = Bh + (long)(k0 + kk) * ldb + tile_n + nc;
#pragma unroll
            for (int j = 0; j < 8; ++j) Bs[buf][(nc + j) * BK + kk] = srcp[j];
        }
    };

    v8f acc[MF][NF];
#pragma unroll
    for (int mf = 0; mf < MF; ++mf)
#pragma unroll
        for (int nf = 0; nf < NF; ++nf)
#pragma unroll
            for (int r = 0; r < 8; ++r) acc[mf][nf][r] = 0.0f;

    // prologue: stage tile 0 into buffer 0
    stageA(0, 0);
    stageB(0, 0);

    int cur = 0;
    for (int k0 = 0; k0 < K; k0 += BK) {
        if (ASY) wait_async0();      // drain this tile's DMA (issued last iter)
        __syncthreads();             // tile `cur` visible; prev reads retired
        if (k0 + BK < K) {           // overlap: DMA tile k+1 during compute
            stageA(1 - cur, k0 + BK);
            stageB(1 - cur, k0 + BK);
        }

        // ---- fragments (per-lane K map of the 16-bit WMMA operand layout) ----
        v16h afrag[MF], bfrag[NF];
#pragma unroll
        for (int mf = 0; mf < MF; ++mf) {
            const int mrow = (w_m0 + mf * 16 + lane16) * BK;
#pragma unroll
            for (int e = 0; e < 16; ++e)
                afrag[mf][e] = As[cur][mrow + ((e < 8) ? e : (e + 8)) + laneHi8];
        }
#pragma unroll
        for (int nf = 0; nf < NF; ++nf) {
            const int nrow = (w_n0 + nf * 16 + lane16) * BK;
#pragma unroll
            for (int e = 0; e < 16; ++e)
                bfrag[nf][e] = Bs[cur][nrow + ((e < 8) ? e : (e + 8)) + laneHi8];
        }

#pragma unroll
        for (int mf = 0; mf < MF; ++mf)
#pragma unroll
            for (int nf = 0; nf < NF; ++nf)
                acc[mf][nf] = __builtin_amdgcn_wmma_f32_16x16x32_f16(
                    false, afrag[mf], false, bfrag[nf],
                    (short)0, acc[mf][nf], false, false);
        cur ^= 1;
    }

    // ---- epilogue: alpha, bias, relu; f16 or f32 store ----
#pragma unroll
    for (int mf = 0; mf < MF; ++mf) {
#pragma unroll
        for (int nf = 0; nf < NF; ++nf) {
            const int n = tile_n + w_n0 + nf * 16 + lane16;
#pragma unroll
            for (int r = 0; r < 8; ++r) {
                const int m = tile_m + w_m0 + mf * 16 + r + laneHi8;
                float v = alpha * acc[mf][nf][r];
                if (bias) v += bias[n];
                if (RELU) v = fmaxf(v, 0.0f);
                if (CH) Chp[(long)m * ldc + n] = (_Float16)v;
                else    Cf[(long)m * ldc + n] = v;
            }
        }
    }
}

// f32 -> f16 bulk convert (n multiple of 4)
__global__ __launch_bounds__(256) void cvt_f32_f16(
    const float* __restrict__ in, _Float16* __restrict__ out, long n)
{
    const long i = ((long)blockIdx.x * 256 + threadIdx.x) * 4;
    if (i + 3 < n) {
        const float4 f = *(const float4*)(in + i);
        out[i]     = (_Float16)f.x;
        out[i + 1] = (_Float16)f.y;
        out[i + 2] = (_Float16)f.z;
        out[i + 3] = (_Float16)f.w;
    }
}

// ---------------------------------------------------------------------------
// Sparse-row compaction (wave32 ballot scan): stable nonzero-first order,
// sentinel token at position cnt, zero-fill beyond.
// ---------------------------------------------------------------------------
__global__ __launch_bounds__(32) void compact_rows(
    const float* __restrict__ src, int* __restrict__ idx, float* __restrict__ w)
{
    const int b = blockIdx.x;
    const int lane = threadIdx.x;
    const float* row = src + (long)b * 2001;
    int cnt = 0;
    for (int base = 0; base < 2001; base += 32) {
        const int c = base + lane;
        float v = (c < 2001) ? row[c] : 0.0f;
        const bool nz = (v != 0.0f);
        const unsigned mask = (unsigned)__ballot(nz);
        const int off = __popc(mask & ((1u << lane) - 1u));
        const int p = cnt + off;
        if (nz && p < 512) { idx[b * 512 + p] = c; w[b * 512 + p] = v; }
        cnt += __popc(mask);
    }
    for (int p = lane; p < 512; p += 32) {
        if (p >= cnt) {
            idx[b * 512 + p] = (p == cnt) ? 2001 : 0;
            w[b * 512 + p]   = (p == cnt) ? 1.0f : 0.0f;
        }
    }
}

// x[b,s,:] = w * emb[idx] * sqrt(512) + PE(s,:)
__global__ __launch_bounds__(256) void embed_pe(
    const float* __restrict__ emb, const int* __restrict__ idx,
    const float* __restrict__ w, float* __restrict__ x)
{
    const long bs = blockIdx.x;
    const int sPos = (int)(bs & 511);
    const int t = threadIdx.x;
    const int id = idx[bs];
    const float wt = w[bs];
    const float* e = emb + (long)id * 512;
    float* xp = x + bs * 512;
    const float scale = 22.62741699796952f;        // sqrt(512)
    const float c = 9.210340371976184f / 512.0f;   // ln(1e4)/512
#pragma unroll
    for (int j = 0; j < 2; ++j) {
        const int d = t + j * 256;
        const float freq = __expf(-(float)(d & ~1) * c);
        const float ang = (float)sPos * freq;
        const float pe = (d & 1) ? __cosf(ang) : __sinf(ang);
        xp[d] = wt * e[d] * scale + pe;
    }
}

// In-place softmax over rows of 512 (1/sqrt(hd) folded into GEMM alpha)
__global__ __launch_bounds__(256) void softmax512(float* __restrict__ data)
{
    __shared__ float red[256];
    float* p = data + (long)blockIdx.x * 512;
    const int t = threadIdx.x;
    const float a = p[t], b = p[t + 256];
    red[t] = fmaxf(a, b);
    __syncthreads();
    for (int s = 128; s > 0; s >>= 1) { if (t < s) red[t] = fmaxf(red[t], red[t + s]); __syncthreads(); }
    const float mx = red[0];
    __syncthreads();
    const float ea = __expf(a - mx), eb = __expf(b - mx);
    red[t] = ea + eb;
    __syncthreads();
    for (int s = 128; s > 0; s >>= 1) { if (t < s) red[t] += red[t + s]; __syncthreads(); }
    const float inv = 1.0f / red[0];
    p[t] = ea * inv;
    p[t + 256] = eb * inv;
}

// x = LayerNorm(x + y) * s + b, rows of 512, in place on x
__global__ __launch_bounds__(256) void add_ln512(
    float* __restrict__ x, const float* __restrict__ y,
    const float* __restrict__ s, const float* __restrict__ b)
{
    __shared__ float red[256];
    float* xp = x + (long)blockIdx.x * 512;
    const float* yp = y + (long)blockIdx.x * 512;
    const int t = threadIdx.x;
    const float a0 = xp[t] + yp[t];
    const float a1 = xp[t + 256] + yp[t + 256];
    red[t] = a0 + a1;
    __syncthreads();
    for (int st = 128; st > 0; st >>= 1) { if (t < st) red[t] += red[t + st]; __syncthreads(); }
    const float mean = red[0] * (1.0f / 512.0f);
    __syncthreads();
    const float d0 = a0 - mean, d1 = a1 - mean;
    red[t] = d0 * d0 + d1 * d1;
    __syncthreads();
    for (int st = 128; st > 0; st >>= 1) { if (t < st) red[t] += red[t + st]; __syncthreads(); }
    const float rstd = rsqrtf(red[0] * (1.0f / 512.0f) + 1e-5f);
    xp[t]       = d0 * rstd * s[t] + b[t];
    xp[t + 256] = d1 * rstd * s[t + 256] + b[t + 256];
}

// y[b,d] = mean_s x[b,s,d]
__global__ __launch_bounds__(512) void mean_seq(
    const float* __restrict__ x, float* __restrict__ y)
{
    const int b = blockIdx.x, d = threadIdx.x;
    const float* p = x + (long)b * 512 * 512 + d;
    float s = 0.0f;
    for (int i = 0; i < 512; ++i) s += p[(long)i * 512];
    y[b * 512 + d] = s * (1.0f / 512.0f);
}

// out[b,o] = y[b,:] . Wl[o,:] + bl[o]   (64 x 10)
__global__ __launch_bounds__(256) void final_linear(
    const float* __restrict__ y, const float* __restrict__ Wl,
    const float* __restrict__ bl, float* __restrict__ out)
{
    const int t = blockIdx.x * blockDim.x + threadIdx.x;
    if (t >= 640) return;
    const int b = t / 10, o = t % 10;
    const float* yp = y + (long)b * 512;
    const float* wp = Wl + (long)o * 512;
    float s = bl[o];
    for (int i = 0; i < 512; ++i) s += yp[i] * wp[i];
    out[t] = s;
}

// ---------------------------------------------------------------------------
extern "C" void kernel_launch(void* const* d_in, const int* in_sizes, int n_in,
                              void* d_out, int out_size, void* d_ws, size_t ws_size,
                              hipStream_t stream)
{
    const float* src  = (const float*)d_in[0];
    const float* emb  = (const float*)d_in[1];
    const float* Wqkv = (const float*)d_in[2];
    const float* bqkv = (const float*)d_in[3];
    const float* Wo   = (const float*)d_in[4];
    const float* bo   = (const float*)d_in[5];
    const float* ln1s = (const float*)d_in[6];
    const float* ln1b = (const float*)d_in[7];
    const float* W1   = (const float*)d_in[8];
    const float* b1   = (const float*)d_in[9];
    const float* W2   = (const float*)d_in[10];
    const float* b2   = (const float*)d_in[11];
    const float* ln2s = (const float*)d_in[12];
    const float* ln2b = (const float*)d_in[13];
    const float* Wl   = (const float*)d_in[14];
    const float* bl   = (const float*)d_in[15];

    const int S = 512, D = 512, B = 64, H = 8, HD = 64, DH = 2048;
    char* ws = (char*)d_ws;
    const size_t MB = (size_t)1 << 20;
    float*     x     = (float*)    (ws);              //  64 MB [B,S,D] f32
    _Float16*  qkvh  = (_Float16*) (ws + 64  * MB);   //  96 MB [B,S,3D] f16
    _Float16*  attnh = (_Float16*) (ws + 160 * MB);   //  32 MB [B,S,D] f16
    float*     proj  = (float*)    (ws + 192 * MB);   //  64 MB sublayer out f32
    float*     scr   = (float*)    (ws + 256 * MB);   //  64 MB scores chunk f32
    _Float16*  scrh  = (_Float16*) (ws + 320 * MB);   //  32 MB ffn hidden chunk f16
    _Float16*  wqh   = (_Float16*) (ws + 352 * MB);   //   3 MB
    _Float16*  woh   = (_Float16*) (ws + 355 * MB);   //   1 MB
    _Float16*  w1h   = (_Float16*) (ws + 356 * MB);   //   4 MB
    _Float16*  w2h   = (_Float16*) (ws + 360 * MB);   //   4 MB
    float*     ymn   = (float*)    (ws + 364 * MB);   // 128 KB
    int*       idx   = (int*)      (ws + 365 * MB);   // 128 KB
    float*     wv    = (float*)    (ws + 366 * MB);   // 128 KB

    // one-time f16 weight images
    cvt_f32_f16<<<(2 * 3 * D * D) / 1024, 256, 0, stream>>>(Wqkv, wqh, 2L * 3 * D * D);
    cvt_f32_f16<<<(2 * D * D) / 1024, 256, 0, stream>>>(Wo, woh, 2L * D * D);
    cvt_f32_f16<<<(2 * DH * D) / 1024, 256, 0, stream>>>(W1, w1h, 2L * DH * D);
    cvt_f32_f16<<<(2 * D * DH) / 1024, 256, 0, stream>>>(W2, w2h, 2L * D * DH);

    compact_rows<<<B, 32, 0, stream>>>(src, idx, wv);
    embed_pe<<<B * S, 256, 0, stream>>>(emb, idx, wv, x);

    for (int l = 0; l < 2; ++l) {
        // QKV: x(f32) @ Wqkv^T -> qkvh (f16)
        {
            dim3 g((3 * D) / 128, (B * S) / BM, 1);
            wmma_gemm<128, false, true, false, true><<<g, 256, 0, stream>>>(
                (const void*)x, D, 0, 0,
                (const void*)(wqh + (long)l * 3 * D * D), D, 0, 0,
                bqkv + (long)l * 3 * D,
                (void*)qkvh, 3 * D, 0, 0,
                B * S, 3 * D, D, 1.0f, 1);
        }
        // attention in chunks of 8 batch elems
        for (int cb = 0; cb < 8; ++cb) {
            const _Float16* qb = qkvh + (long)(cb * 8) * S * (3 * D);
            {   // scores = 0.125 * q @ k^T  (both f16, async-staged)
                dim3 g(S / 128, S / BM, 64);
                wmma_gemm<128, true, true, false, false><<<g, 256, 0, stream>>>(
                    (const void*)qb,       3 * D, (long)S * 3 * D, HD,
                    (const void*)(qb + D), 3 * D, (long)S * 3 * D, HD,
                    nullptr,
                    (void*)scr, S, (long)H * S * S, (long)S * S,
                    S, S, HD, 0.125f, H);
            }
            softmax512<<<64 * S, 256, 0, stream>>>(scr);
            {   // o = probs(f32) @ v(f16,[K,N]) -> attnh (f16)
                dim3 g(1, S / BM, 64);
                wmma_gemm<64, false, false, false, true><<<g, 256, 0, stream>>>(
                    (const void*)scr, S, (long)H * S * S, (long)S * S,
                    (const void*)(qb + 2 * D), 3 * D, (long)S * 3 * D, HD,
                    nullptr,
                    (void*)(attnh + (long)(cb * 8) * S * D), D, (long)S * D, HD,
                    S, HD, S, 1.0f, H);
            }
        }
        // output projection: attnh(f16) @ Wo^T -> proj (f32)
        {
            dim3 g(D / 128, (B * S) / BM, 1);
            wmma_gemm<128, true, true, false, false><<<g, 256, 0, stream>>>(
                (const void*)attnh, D, 0, 0,
                (const void*)(woh + (long)l * D * D), D, 0, 0,
                bo + (long)l * D,
                (void*)proj, D, 0, 0,
                B * S, D, D, 1.0f, 1);
        }
        add_ln512<<<B * S, 256, 0, stream>>>(x, proj, ln1s + l * D, ln1b + l * D);

        // FFN in 4 row-chunks of 8192
        for (int mc = 0; mc < 4; ++mc) {
            const float* xa = x + (long)mc * 8192 * D;
            {   // up: x(f32) @ W1^T, ReLU -> scrh (f16)
                dim3 g(DH / 128, 8192 / BM, 1);
                wmma_gemm<128, false, true, true, true><<<g, 256, 0, stream>>>(
                    (const void*)xa, D, 0, 0,
                    (const void*)(w1h + (long)l * DH * D), D, 0, 0,
                    b1 + (long)l * DH,
                    (void*)scrh, DH, 0, 0,
                    8192, DH, D, 1.0f, 1);
            }
            {   // down: scrh(f16) @ W2^T -> proj (f32)
                dim3 g(D / 128, 8192 / BM, 1);
                wmma_gemm<128, true, true, false, false><<<g, 256, 0, stream>>>(
                    (const void*)scrh, DH, 0, 0,
                    (const void*)(w2h + (long)l * D * DH), DH, 0, 0,
                    b2 + (long)l * D,
                    (void*)(proj + (long)mc * 8192 * D), D, 0, 0,
                    8192, D, DH, 1.0f, 1);
            }
        }
        add_ln512<<<B * S, 256, 0, stream>>>(x, proj, ln2s + l * D, ln2b + l * D);
    }

    mean_seq<<<B, 512, 0, stream>>>(x, ymn);
    final_linear<<<3, 256, 0, stream>>>(ymn, Wl, bl, (float*)d_out);
}
